// FullChiralModel_11982958756600
// MI455X (gfx1250) — compile-verified, matrix operand order checked by reading, not run on hardware
//
#include <hip/hip_runtime.h>
#include <cstdint>
#include <cstddef>

// ---------------------------------------------------------------------------
// Constants from the reference model
// ---------------------------------------------------------------------------
#define DF   128      // feature dim
#define RNUM 8        // relations
#define GNUM 64       // graphs
#define CNUM 10       // classes
#define NB   2        // bases

// ---------------------------------------------------------------------------
// CDNA5 WMMA types (wave32; 16x16x32 bf16 -> f32)
// ---------------------------------------------------------------------------
typedef __attribute__((ext_vector_type(16))) __bf16 v16bf;
typedef __attribute__((ext_vector_type(8)))  float  v8f;

union Frag { v16bf v; unsigned u[8]; };

__device__ __forceinline__ unsigned short f32_to_bf16(float f) {
  union { float f; unsigned u; } c; c.f = f;
  unsigned u = c.u;
  if ((u & 0x7f800000u) == 0x7f800000u) return (unsigned short)(u >> 16); // inf/nan
  unsigned r = u + 0x7fffu + ((u >> 16) & 1u);                            // RNE
  return (unsigned short)(r >> 16);
}

// A fragment: 16x32 bf16.  ISA 7.12.2: lanes 0-15 hold M=lane; VGPR v holds
// K = (v<4?0:16) + (lane>=16?8:0) + 2*(v&3) + half.  Adjacent K pairs -> u32.
__device__ __forceinline__ v16bf load_a_frag(const unsigned short* As, int lane) {
  Frag f;
  int m   = lane & 15;
  int hi8 = (lane & 16) ? 8 : 0;
#pragma unroll
  for (int v = 0; v < 8; ++v) {
    int kb = ((v < 4) ? 0 : 16) + hi8 + (v & 3) * 2;
    f.u[v] = *(const unsigned*)(As + m * 32 + kb);
  }
  return f.v;
}
// B fragment: 32x16 bf16, mirrored layout (lane = N column, same K interleave).
// Bt is staged transposed in LDS as [n][k] so the K pair is one u32.
__device__ __forceinline__ v16bf load_b_frag(const unsigned short* Bt, int lane, int ntile) {
  Frag f;
  int n   = ntile * 16 + (lane & 15);
  int hi8 = (lane & 16) ? 8 : 0;
#pragma unroll
  for (int v = 0; v < 8; ++v) {
    int kb = ((v < 4) ? 0 : 16) + hi8 + (v & 3) * 2;
    f.u[v] = *(const unsigned*)(Bt + n * 32 + kb);
  }
  return f.v;
}

// ---------------------------------------------------------------------------
// WMMA GEMM: C[M,N] = A[M,K](f32) @ Wb[K,N](bf16) + bias, optional relu.
// Block = 8 waves, 128-row tile; wave w owns rows 16w..16w+15; NT = N/16.
// ---------------------------------------------------------------------------
template <int NT>
__global__ __launch_bounds__(256) void k_gemm_bf16(
    const float* __restrict__ A, const unsigned short* __restrict__ Wb,
    const float* __restrict__ bias, float* __restrict__ C, int M, int K, int relu) {
  constexpr int N = NT * 16;
  __shared__ unsigned short As[128 * 32];   // [row][k]
  __shared__ unsigned short Bs[128 * 32];   // [n][k] (transposed)
  int tid = threadIdx.x, wave = tid >> 5, lane = tid & 31;
  int row0 = blockIdx.x * 128;

  v8f acc[NT];
#pragma unroll
  for (int t = 0; t < NT; ++t) {
#pragma unroll
    for (int i = 0; i < 8; ++i) acc[t][i] = 0.0f;
  }

  for (int k0 = 0; k0 < K; k0 += 32) {
    __syncthreads();
    { // stage A tile (128 x 32) fp32 -> bf16
      int r  = tid >> 1;
      int cb = (tid & 1) * 16;
      unsigned short* dstp = &As[r * 32 + cb];
      if (row0 + r < M) {
        const float* srcp = A + (size_t)(row0 + r) * K + k0 + cb;
        if (k0 + 32 < K) __builtin_prefetch(srcp + 32, 0, 3);  // global_prefetch_b8
#pragma unroll
        for (int i = 0; i < 16; ++i) dstp[i] = f32_to_bf16(srcp[i]);
      } else {
#pragma unroll
        for (int i = 0; i < 16; ++i) dstp[i] = 0;
      }
    }
    // stage B tile (32 x N) transposed into Bs[n][k]
    for (int e = tid; e < 32 * N; e += 256) {
      int k = e / N, n = e % N;
      Bs[n * 32 + k] = Wb[(size_t)(k0 + k) * N + n];
    }
    __syncthreads();

    const unsigned short* Aw = &As[wave * 16 * 32];
    v16bf af = load_a_frag(Aw, lane);
#pragma unroll
    for (int t = 0; t < NT; ++t) {
      v16bf bfv = load_b_frag(Bs, lane, t);
      acc[t] = __builtin_amdgcn_wmma_f32_16x16x32_bf16(
          false, af, false, bfv, (short)0, acc[t], false, false);
    }
  }

  // epilogue: C/D layout (VGPR v: M=v lanes0-15, M=v+8 lanes16-31; N=lane%16)
  int m_off = (lane & 16) ? 8 : 0;
  int col0  = lane & 15;
#pragma unroll
  for (int t = 0; t < NT; ++t) {
#pragma unroll
    for (int v = 0; v < 8; ++v) {
      int row = row0 + wave * 16 + v + m_off;
      int col = t * 16 + col0;
      if (row < M) {
        float val = acc[t][v] + (bias ? bias[col] : 0.0f);
        if (relu) val = fmaxf(val, 0.0f);
        C[(size_t)row * N + col] = val;
      }
    }
  }
}

// ---------------------------------------------------------------------------
// Small utility kernels
// ---------------------------------------------------------------------------
__global__ void k_fillf(float* p, long n, float v) {
  long i = blockIdx.x * (long)blockDim.x + threadIdx.x; if (i < n) p[i] = v;
}
__global__ void k_filli(int* p, long n, int v) {
  long i = blockIdx.x * (long)blockDim.x + threadIdx.x; if (i < n) p[i] = v;
}
__global__ void k_cvt(const float* __restrict__ w, unsigned short* __restrict__ o, long n) {
  long i = blockIdx.x * (long)blockDim.x + threadIdx.x; if (i < n) o[i] = f32_to_bf16(w[i]);
}
__global__ void k_relu(float* p, long n) {
  long i = blockIdx.x * (long)blockDim.x + threadIdx.x; if (i < n) p[i] = fmaxf(p[i], 0.0f);
}
__global__ void k_broadcast_row(const float* __restrict__ row, float* __restrict__ out, int n) {
  long idx = blockIdx.x * (long)blockDim.x + threadIdx.x;
  if (idx < (long)n * DF) out[idx] = row[idx & (DF - 1)];
}
__global__ void k_rowlin(const float* __restrict__ row, const float* __restrict__ W,
                         const float* __restrict__ b, float* __restrict__ out) {
  __shared__ float r[DF];
  int j = threadIdx.x; r[j] = row[j]; __syncthreads();
  float acc = b[j];
  for (int k = 0; k < DF; ++k) acc += r[k] * W[k * DF + j];
  out[j] = acc;
}
__global__ void k_gather_align(const float* __restrict__ in, int ns,
                               float* __restrict__ out, int nt) {
  long idx = blockIdx.x * (long)blockDim.x + threadIdx.x;
  if (idx >= (long)nt * DF) return;
  long i = idx >> 7; int j = (int)(idx & (DF - 1));
  long si = i * (long)ns / nt; if (si > ns - 1) si = ns - 1;
  out[idx] = in[si * DF + j];
}

// ---------------------------------------------------------------------------
// R-GCN edge kernels (wave-per-edge, D=128 -> 4 floats/lane)
// ---------------------------------------------------------------------------
__global__ void k_deg(const int* __restrict__ d, const int* __restrict__ t,
                      const float* __restrict__ m, float* __restrict__ deg, int E_) {
  int e = blockIdx.x * blockDim.x + threadIdx.x; if (e >= E_) return;
  float em = m ? m[e] : 1.0f;
  if (em != 0.0f) atomicAdd(&deg[(size_t)d[e] * RNUM + t[e]], em);
}

__global__ __launch_bounds__(256) void k_edge_msg(
    const int* __restrict__ s, const int* __restrict__ d, const int* __restrict__ t,
    const float* __restrict__ m, const float* __restrict__ x,
    const float* __restrict__ hb0, const float* __restrict__ hb1,
    const float* __restrict__ comp, const float* __restrict__ cws,
    const float* __restrict__ cwd, const float* __restrict__ cb,
    const float* __restrict__ deg, float* __restrict__ out, int E_) {
  int wid  = (int)((blockIdx.x * (long)blockDim.x + threadIdx.x) >> 5);
  int lane = threadIdx.x & 31;
  if (wid >= E_) return;
  float em = m ? m[wid] : 1.0f;
  if (em == 0.0f) return;
  int si = s[wid], di = d[wid], ti = t[wid];
  float4 a  = ((const float4*)(x + (size_t)si * DF))[lane];
  float4 b  = ((const float4*)(x + (size_t)di * DF))[lane];
  float4 ws = ((const float4*)cws)[lane];
  float4 wd = ((const float4*)cwd)[lane];
  float p = a.x * ws.x + a.y * ws.y + a.z * ws.z + a.w * ws.w +
            b.x * wd.x + b.y * wd.y + b.z * wd.z + b.w * wd.w;
#pragma unroll
  for (int o = 16; o > 0; o >>= 1) p += __shfl_xor(p, o, 32);
  float conf = 1.0f / (1.0f + __expf(-(p + cb[0])));
  float w = conf * em / fmaxf(deg[(size_t)di * RNUM + ti], 1.0f);
  float c0 = comp[ti * NB + 0], c1 = comp[ti * NB + 1];
  float4 h0 = ((const float4*)(hb0 + (size_t)si * DF))[lane];
  float4 h1 = ((const float4*)(hb1 + (size_t)si * DF))[lane];
  float* o4 = out + (size_t)di * DF + lane * 4;
  atomicAdd(o4 + 0, (c0 * h0.x + c1 * h1.x) * w);
  atomicAdd(o4 + 1, (c0 * h0.y + c1 * h1.y) * w);
  atomicAdd(o4 + 2, (c0 * h0.z + c1 * h1.z) * w);
  atomicAdd(o4 + 3, (c0 * h0.w + c1 * h1.w) * w);
}

// ---------------------------------------------------------------------------
// Top-K pooling: score + 4-pass radix select + compaction
// ---------------------------------------------------------------------------
__global__ void k_vecnorm(const float* __restrict__ p, float* __restrict__ out) {
  __shared__ float sm[DF];
  int t = threadIdx.x; float v = p[t]; sm[t] = v * v; __syncthreads();
  for (int s = 64; s > 0; s >>= 1) { if (t < s) sm[t] += sm[t + s]; __syncthreads(); }
  if (t == 0) out[0] = sqrtf(sm[0]);
}
__global__ __launch_bounds__(256) void k_score(
    const float* __restrict__ x, const float* __restrict__ p, const float* __restrict__ pn,
    float* __restrict__ score, unsigned* __restrict__ keys, int n) {
  int wid = (int)((blockIdx.x * (long)blockDim.x + threadIdx.x) >> 5);
  int lane = threadIdx.x & 31;
  if (wid >= n) return;
  float4 a = ((const float4*)(x + (size_t)wid * DF))[lane];
  float4 b = ((const float4*)p)[lane];
  float s = a.x * b.x + a.y * b.y + a.z * b.z + a.w * b.w;
#pragma unroll
  for (int o = 16; o > 0; o >>= 1) s += __shfl_xor(s, o, 32);
  if (lane == 0) {
    float sc = s / (pn[0] + 1e-16f);
    score[wid] = sc;
    unsigned u = __float_as_uint(sc);
    keys[wid] = (u & 0x80000000u) ? ~u : (u | 0x80000000u);  // order-preserving key
  }
}
__global__ void k_radix_init(unsigned* st, int K) {
  if (threadIdx.x == 0) { st[0] = 0u; st[1] = 0u; st[2] = (unsigned)K; st[3] = 0u; st[4] = 0u; }
}
__global__ void k_hist(const unsigned* __restrict__ keys, int n, const unsigned* __restrict__ st,
                       int* __restrict__ hist, int shift) {
  int i = blockIdx.x * blockDim.x + threadIdx.x; if (i >= n) return;
  unsigned k = keys[i];
  if ((k & st[1]) == st[0]) atomicAdd(&hist[(k >> shift) & 255], 1);
}
__global__ void k_pick(const int* __restrict__ hist, unsigned* st, int shift, int last, int K) {
  if (threadIdx.x != 0 || blockIdx.x != 0) return;
  int rem = (int)st[2];
  for (int d = 255; d >= 0; --d) {
    int c = hist[d];
    if (rem <= c) { st[0] |= ((unsigned)d) << shift; st[1] |= 255u << shift; st[2] = (unsigned)rem; break; }
    rem -= c;
  }
  if (last) { st[3] = 0u; st[4] = (unsigned)(K - (int)st[2]); }  // tie region starts at count_gt
}
__global__ void k_compact(const unsigned* __restrict__ keys, unsigned* st,
                          int* __restrict__ perm, int* __restrict__ newidx, int n, int K) {
  int i = blockIdx.x * blockDim.x + threadIdx.x; if (i >= n) return;
  unsigned k = keys[i], T = st[0];
  int pos = -1;
  if (k > T) pos = (int)atomicAdd(&st[3], 1u);
  else if (k == T) { unsigned p = atomicAdd(&st[4], 1u); if (p < (unsigned)K) pos = (int)p; }
  if (pos >= 0) { perm[pos] = i; newidx[i] = pos; }
}
__global__ __launch_bounds__(256) void k_gather_scale(
    const float* __restrict__ x, const int* __restrict__ bin, const int* __restrict__ perm,
    const float* __restrict__ score, float* __restrict__ xout, int* __restrict__ bout, int K) {
  int wid = (int)((blockIdx.x * (long)blockDim.x + threadIdx.x) >> 5);
  int lane = threadIdx.x & 31;
  if (wid >= K) return;
  int pi = perm[wid];
  float g = tanhf(score[pi]);
  float4 a = ((const float4*)(x + (size_t)pi * DF))[lane];
  float4 r; r.x = a.x * g; r.y = a.y * g; r.z = a.z * g; r.w = a.w * g;
  ((float4*)(xout + (size_t)wid * DF))[lane] = r;
  if (lane == 0) bout[wid] = bin[pi];
}
__global__ void k_remap(const int* __restrict__ newidx, const int* __restrict__ s,
                        const int* __restrict__ d, const float* __restrict__ mIn,
                        int* __restrict__ sO, int* __restrict__ dO, float* __restrict__ mO, int E_) {
  int e = blockIdx.x * blockDim.x + threadIdx.x; if (e >= E_) return;
  int ns = newidx[s[e]], nd = newidx[d[e]];
  float mi = mIn ? mIn[e] : 1.0f;
  bool keep = (ns >= 0) && (nd >= 0) && (mi > 0.0f);
  sO[e] = keep ? ns : 0; dO[e] = keep ? nd : 0; mO[e] = keep ? 1.0f : 0.0f;
}

// ---------------------------------------------------------------------------
// norm01 stats + hinge pieces
// ---------------------------------------------------------------------------
__global__ void k_colminmax(const float* __restrict__ x, int n,
                            float* __restrict__ mn, float* __restrict__ sc) {
  __shared__ float smn[256], smx[256];
  int j = blockIdx.x, t = threadIdx.x;
  float lo = 3.4e38f, hi = -3.4e38f;
  for (int i = t; i < n; i += 256) {
    float v = x[(size_t)i * DF + j];
    lo = fminf(lo, v); hi = fmaxf(hi, v);
  }
  smn[t] = lo; smx[t] = hi; __syncthreads();
  for (int s = 128; s > 0; s >>= 1) {
    if (t < s) { smn[t] = fminf(smn[t], smn[t + s]); smx[t] = fmaxf(smx[t], smx[t + s]); }
    __syncthreads();
  }
  if (t == 0) { float d = smx[0] - smn[0]; mn[j] = smn[0]; sc[j] = (d < 1e-8f) ? 1.0f : d; }
}
// Fold min-max normalization into the linear layer:  W'=W/sc ,  b'=b-(mn/sc)@W
__global__ void k_prep_normW(const float* __restrict__ W, const float* __restrict__ b,
                             const float* __restrict__ mn, const float* __restrict__ sc,
                             unsigned short* __restrict__ Wb, float* __restrict__ bp) {
  int j = threadIdx.x;
  float acc = b[j];
  for (int k = 0; k < DF; ++k) {
    float w = W[k * DF + j], is = 1.0f / sc[k];
    Wb[k * DF + j] = f32_to_bf16(w * is);
    acc -= mn[k] * is * w;
  }
  bp[j] = acc;
}
__device__ __forceinline__ float gelu_exact(float y) {
  return 0.5f * y * (1.0f + erff(y * 0.70710678118654752f));
}
__global__ __launch_bounds__(256) void k_ln_gelu(
    const float* __restrict__ z, const float* __restrict__ g, const float* __restrict__ b2,
    float* __restrict__ out, int n) {
  int wid = (int)((blockIdx.x * (long)blockDim.x + threadIdx.x) >> 5);
  int lane = threadIdx.x & 31;
  if (wid >= n) return;
  float4 v = ((const float4*)(z + (size_t)wid * DF))[lane];
  float s = v.x + v.y + v.z + v.w;
  float q = v.x * v.x + v.y * v.y + v.z * v.z + v.w * v.w;
#pragma unroll
  for (int o = 16; o > 0; o >>= 1) { s += __shfl_xor(s, o, 32); q += __shfl_xor(q, o, 32); }
  float mu = s * (1.0f / DF);
  float var = q * (1.0f / DF) - mu * mu;
  float inv = rsqrtf(var + 1e-5f);
  float4 gg = ((const float4*)g)[lane], bb = ((const float4*)b2)[lane];
  float4 r;
  r.x = gelu_exact((v.x - mu) * inv * gg.x + bb.x);
  r.y = gelu_exact((v.y - mu) * inv * gg.y + bb.y);
  r.z = gelu_exact((v.z - mu) * inv * gg.z + bb.z);
  r.w = gelu_exact((v.w - mu) * inv * gg.w + bb.w);
  ((float4*)(out + (size_t)wid * DF))[lane] = r;
}
// out = a*(xraw-mn) + (1-a)*lt*sc + mn     (a = sigmoid(gate[j]))
__global__ void k_hinge_fuse(const float* __restrict__ xraw, const float* __restrict__ lt,
                             const float* __restrict__ mn, const float* __restrict__ sc,
                             const float* __restrict__ gate, float* __restrict__ out, int n) {
  long idx = blockIdx.x * (long)blockDim.x + threadIdx.x;
  if (idx >= (long)n * DF) return;
  int j = (int)(idx & (DF - 1));
  float a = 1.0f / (1.0f + __expf(-gate[j]));
  out[idx] = a * (xraw[idx] - mn[j]) + (1.0f - a) * lt[idx] * sc[j] + mn[j];
}

// ---------------------------------------------------------------------------
// seg_mean / MLP head / MSE
// ---------------------------------------------------------------------------
__global__ void k_seg_count(const int* __restrict__ seg, float* __restrict__ cnt, int n) {
  int i = blockIdx.x * blockDim.x + threadIdx.x;
  if (i < n) atomicAdd(&cnt[seg[i]], 1.0f);
}
__global__ void k_seg_accum(const float* __restrict__ z, const int* __restrict__ seg,
                            float* __restrict__ sum, long n128) {
  long idx = blockIdx.x * (long)blockDim.x + threadIdx.x;
  if (idx >= n128) return;
  int i = (int)(idx >> 7), j = (int)(idx & (DF - 1));
  atomicAdd(&sum[(size_t)seg[i] * DF + j], z[idx]);
}
__global__ void k_seg_div(const float* __restrict__ sum, const float* __restrict__ cnt,
                          float* __restrict__ out, int gd) {
  int idx = blockIdx.x * blockDim.x + threadIdx.x;
  if (idx < gd) out[idx] = sum[idx] / fmaxf(cnt[idx >> 7], 1.0f);
}
__global__ void k_mlp2_acc(const float* __restrict__ z, const float* __restrict__ W2,
                           const float* __restrict__ b2, float* __restrict__ out, int G_) {
  int idx = blockIdx.x * blockDim.x + threadIdx.x;
  if (idx >= G_ * CNUM) return;
  int g = idx / CNUM, c = idx % CNUM;
  float acc = b2[c];
  for (int k = 0; k < 64; ++k) acc += z[g * 64 + k] * W2[k * CNUM + c];
  atomicAdd(&out[g * CNUM + c], acc * (1.0f / 3.0f));
}
__global__ __launch_bounds__(256) void k_mse(const float* __restrict__ a,
                                             const float* __restrict__ b,
                                             float* __restrict__ acc, long n) {
  __shared__ float sm[256];
  float s = 0.0f;
  for (long i = blockIdx.x * (long)blockDim.x + threadIdx.x; i < n;
       i += (long)gridDim.x * blockDim.x) {
    float d = a[i] - b[i]; s += d * d;
  }
  sm[threadIdx.x] = s; __syncthreads();
  for (int o = 128; o > 0; o >>= 1) { if (threadIdx.x < o) sm[threadIdx.x] += sm[threadIdx.x + o]; __syncthreads(); }
  if (threadIdx.x == 0) atomicAdd(acc, sm[0]);
}
__global__ void k_final(const float* __restrict__ accs, float* __restrict__ out, float scale) {
  if (threadIdx.x == 0 && blockIdx.x == 0) {
    out[GNUM * CNUM + 0] = accs[0] / scale;
    out[GNUM * CNUM + 1] = accs[1] / scale;
    out[GNUM * CNUM + 2] = 0.5f * (accs[2] + accs[3]) / scale;
  }
}

// ---------------------------------------------------------------------------
// Host orchestration
//
// d_in ordering = jax pytree flatten (dict keys sorted alphabetically, DFS):
//  0 batch, 1 edge_index, 2 edge_type, 3.. params (see indices below), last x.
// ---------------------------------------------------------------------------
extern "C" void kernel_launch(void* const* d_in, const int* in_sizes, int n_in,
                              void* d_out, int out_size, void* d_ws, size_t ws_size,
                              hipStream_t stream) {
  (void)out_size; (void)ws_size;
  const int N  = in_sizes[0];
  const int E  = in_sizes[2];
  const int K2 = N / 2, K3 = N / 4;

  const int* batch = (const int*)d_in[0];
  const int* ei    = (const int*)d_in[1];
  const int* et    = (const int*)d_in[2];
  const int* src   = ei;
  const int* dst   = ei + E;
  auto F = [&](int i) -> const float* { return (const float*)d_in[i]; };
  const float* x = F(n_in - 1);  // 'x' sorts last

  // param leaf indices (alphabetical DFS):
  // align_l6_to_l1: b=3 w=4 | hinge_l1_l6: 5..14 | hinge_l2_l5: 15..24 |
  // hinge_l3_l4: 25..34 | l6_prior=35 | pool1_p=36 | pool2_p=37 |
  // pred_l1: 38..41 | pred_l2: 42..45 | pred_l3: 46..49 |
  // rec_l1_from_l3: 50,51 | rec_l1_from_l6: 52,53 | rec_l6_from_l1: 54,55 |
  // rec_l6_from_l4: 56,57 | rgcn_l1: 58..64 | l2: 65..71 | l3: 72..78 |
  // l4: 79..85 | l5: 86..92 | unpool_l5_to_l4: b=93 w=94 | unpool_l6_to_l5: b=95 w=96

  char* wsb = (char*)d_ws; size_t off = 0;
  auto alloc = [&](size_t bytes) -> void* {
    void* p = wsb + off; off = (off + bytes + 255) & ~(size_t)255; return p;
  };
  const size_t ND = (size_t)N * DF * 4, K2D = (size_t)K2 * DF * 4, K3D = (size_t)K3 * DF * 4;
  float* xl1   = (float*)alloc(ND);
  float* xl4   = (float*)alloc(ND);
  float* xl6up = (float*)alloc(ND);
  float* x1ref = (float*)alloc(ND);
  float* x6ref = (float*)alloc(ND);
  float* x3a   = (float*)alloc(ND);
  float* tmpN  = (float*)alloc(ND);   // x_l4_in / z temp / mse pred
  float* hb0   = (float*)alloc(ND);   // basis-0 features / lt temp
  float* hb1   = (float*)alloc(ND);   // basis-1 features / ut temp
  float* x3ref = (float*)alloc(ND);
  float* x4ref = (float*)alloc(ND);
  float* x2    = (float*)alloc(K2D);  // pooled feats; reused as x_l5_in
  float* xl2   = (float*)alloc(K2D);
  float* xl5   = (float*)alloc(K2D);
  float* x2ref = (float*)alloc(K2D);
  float* x3p   = (float*)alloc(K3D);  // pool2 output feats
  float* xl3   = (float*)alloc(K3D);
  int*   s2 = (int*)alloc((size_t)E * 4);  int* d2 = (int*)alloc((size_t)E * 4);
  float* m2 = (float*)alloc((size_t)E * 4);
  int*   s3 = (int*)alloc((size_t)E * 4);  int* d3 = (int*)alloc((size_t)E * 4);
  float* m3 = (float*)alloc((size_t)E * 4);
  float*    deg    = (float*)alloc((size_t)N * RNUM * 4);
  float*    score  = (float*)alloc((size_t)N * 4);
  unsigned* keys   = (unsigned*)alloc((size_t)N * 4);
  int*      newidx = (int*)alloc((size_t)N * 4);
  int*      perm   = (int*)alloc((size_t)K2 * 4);
  int*      b2     = (int*)alloc((size_t)K2 * 4);
  int*      b3     = (int*)alloc((size_t)K3 * 4);
  unsigned short* wb16 = (unsigned short*)alloc(DF * DF * 2);
  float* bprime = (float*)alloc(DF * 4);
  float* rowA   = (float*)alloc(DF * 4);
  float* mnA = (float*)alloc(DF * 4); float* scA = (float*)alloc(DF * 4);
  float* mnB = (float*)alloc(DF * 4); float* scB = (float*)alloc(DF * 4);
  float*    pnorm = (float*)alloc(4);
  int*      hist  = (int*)alloc(256 * 4);
  unsigned* st    = (unsigned*)alloc(8 * 4);
  float* segsum = (float*)alloc((size_t)GNUM * DF * 4);
  float* segcnt = (float*)alloc(GNUM * 4);
  float* segm   = (float*)alloc((size_t)GNUM * DF * 4);
  float* z1     = (float*)alloc((size_t)GNUM * 64 * 4);
  float* accs   = (float*)alloc(4 * 4);

  auto cd = [](long a, long b) -> int { return (int)((a + b - 1) / b); };
  float* outF = (float*)d_out;
  k_fillf<<<cd(GNUM * CNUM + 3, 256), 256, 0, stream>>>(outF, GNUM * CNUM + 3, 0.0f);

  // C = A @ W(f32,[128x128]) + bias, via bf16 WMMA
  auto gemm128 = [&](const float* A, const float* W, const float* bias, float* C, int M) {
    k_cvt<<<cd(DF * DF, 256), 256, 0, stream>>>(W, wb16, DF * DF);
    k_gemm_bf16<8><<<cd(M, 128), 256, 0, stream>>>(A, wb16, bias, C, M, DF, 0);
  };

  auto rgcn = [&](const float* xin, int n, const int* s, const int* d, const int* t,
                  const float* m, int pb, float* out) {
    // pb: bases=+0 bias=+1 cb=+2 comp=+3 cw_dst=+4 cw_src=+5 root=+6
    gemm128(xin, F(pb + 0),            nullptr,   hb0, n);
    gemm128(xin, F(pb + 0) + DF * DF,  nullptr,   hb1, n);
    gemm128(xin, F(pb + 6),            F(pb + 1), out, n);
    k_fillf<<<cd((long)n * RNUM, 256), 256, 0, stream>>>(deg, (long)n * RNUM, 0.0f);
    k_deg<<<cd(E, 256), 256, 0, stream>>>(d, t, m, deg, E);
    k_edge_msg<<<cd(E, 8), 256, 0, stream>>>(s, d, t, m, xin, hb0, hb1,
                                             F(pb + 3), F(pb + 5), F(pb + 4), F(pb + 2),
                                             deg, out, E);
    k_relu<<<cd((long)n * DF, 256), 256, 0, stream>>>(out, (long)n * DF);
  };

  auto topk = [&](const float* xin, int n, const int* s, const int* d, const float* mIn,
                  const int* bIn, const float* pvec, int K, float* xout,
                  int* sO, int* dO, float* mO, int* bO) {
    k_vecnorm<<<1, DF, 0, stream>>>(pvec, pnorm);
    k_score<<<cd(n, 8), 256, 0, stream>>>(xin, pvec, pnorm, score, keys, n);
    k_radix_init<<<1, 1, 0, stream>>>(st, K);
    for (int p = 0; p < 4; ++p) {
      int shift = 24 - 8 * p;
      k_filli<<<1, 256, 0, stream>>>(hist, 256, 0);
      k_hist<<<cd(n, 256), 256, 0, stream>>>(keys, n, st, hist, shift);
      k_pick<<<1, 1, 0, stream>>>(hist, st, shift, (p == 3) ? 1 : 0, K);
    }
    k_filli<<<cd(n, 256), 256, 0, stream>>>(newidx, n, -1);
    k_compact<<<cd(n, 256), 256, 0, stream>>>(keys, st, perm, newidx, n, K);
    k_gather_scale<<<cd(K, 8), 256, 0, stream>>>(xin, bIn, perm, score, xout, bO, K);
    k_remap<<<cd(E, 256), 256, 0, stream>>>(newidx, s, d, mIn, sO, dO, mO, E);
  };

  // normalized Linear -> LN -> GELU (normalization folded into W/b)
  auto hingeTr = [&](const float* X, int n, const float* mn, const float* sc,
                     int bi, int b2i, int gi, int wi, float* z, float* outT) {
    k_prep_normW<<<1, DF, 0, stream>>>(F(wi), F(bi), mn, sc, wb16, bprime);
    k_gemm_bf16<8><<<cd(n, 128), 256, 0, stream>>>(X, wb16, bprime, z, n, DF, 0);
    k_ln_gelu<<<cd(n, 8), 256, 0, stream>>>(z, F(gi), F(b2i), outT, n);
  };

  // ===== Upper trifold =====
  rgcn(x, N, src, dst, et, nullptr, 58, xl1);                                  // L1
  topk(xl1, N, src, dst, nullptr, batch, F(36), K2, x2, s2, d2, m2, b2);       // pool1
  rgcn(x2, K2, s2, d2, et, m2, 65, xl2);                                       // L2
  topk(xl2, K2, s2, d2, m2, b2, F(37), K3, x3p, s3, d3, m3, b3);               // pool2
  rgcn(x3p, K3, s3, d3, et, m3, 72, xl3);                                      // L3

  // ===== Lower trifold (x_l6 is a broadcast prior row) =====
  k_rowlin<<<1, DF, 0, stream>>>(F(35), F(96), F(95), rowA);                   // prior@unpool_l6_to_l5
  k_broadcast_row<<<cd((long)K2 * DF, 256), 256, 0, stream>>>(rowA, x2, K2);   // x_l5_in (reuse x2)
  rgcn(x2, K2, s2, d2, et, m2, 86, xl5);                                       // L5
  k_gather_align<<<cd((long)N * DF, 256), 256, 0, stream>>>(xl5, K2, hb0, N);  // align K2->N
  gemm128(hb0, F(94), F(93), tmpN, N);                                         // x_l4_in
  rgcn(tmpN, N, src, dst, et, nullptr, 79, xl4);                               // L4
  k_rowlin<<<1, DF, 0, stream>>>(F(35), F(4), F(3), rowA);                     // prior@align_l6_to_l1
  k_broadcast_row<<<cd((long)N * DF, 256), 256, 0, stream>>>(rowA, xl6up, N);  // x_l6_up

  // ===== Hinge 1: L1 <-> L6 =====
  k_colminmax<<<DF, 256, 0, stream>>>(xl1,   N, mnA, scA);
  k_colminmax<<<DF, 256, 0, stream>>>(xl6up, N, mnB, scB);
  hingeTr(xl6up, N, mnB, scB, 7, 8, 9, 10, tmpN, hb0);    // lt = t_l4u(n6)
  hingeTr(xl1,   N, mnA, scA, 11, 12, 13, 14, tmpN, hb1); // ut = t_u4l(n1)
  k_hinge_fuse<<<cd((long)N * DF, 256), 256, 0, stream>>>(xl1,   hb0, mnA, scA, F(5), x1ref, N);
  k_hinge_fuse<<<cd((long)N * DF, 256), 256, 0, stream>>>(xl6up, hb1, mnB, scB, F(6), x6ref, N);

  // ===== Hinge 2: L2 <-> L5 (only r2 consumed) =====
  k_colminmax<<<DF, 256, 0, stream>>>(xl2, K2, mnA, scA);
  k_colminmax<<<DF, 256, 0, stream>>>(xl5, K2, mnB, scB);
  hingeTr(xl5, K2, mnB, scB, 17, 18, 19, 20, tmpN, hb0);
  k_hinge_fuse<<<cd((long)K2 * DF, 256), 256, 0, stream>>>(xl2, hb0, mnA, scA, F(15), x2ref, K2);

  // ===== Hinge 3: L3 <-> L4 =====
  k_gather_align<<<cd((long)N * DF, 256), 256, 0, stream>>>(xl3, K3, x3a, N);
  k_colminmax<<<DF, 256, 0, stream>>>(x3a, N, mnA, scA);
  k_colminmax<<<DF, 256, 0, stream>>>(xl4, N, mnB, scB);
  hingeTr(xl4, N, mnB, scB, 27, 28, 29, 30, tmpN, hb0);   // lt = t_l4u(n4)
  hingeTr(x3a, N, mnA, scA, 31, 32, 33, 34, tmpN, hb1);   // ut = t_u4l(n3)
  k_hinge_fuse<<<cd((long)N * DF, 256), 256, 0, stream>>>(x3a, hb0, mnA, scA, F(25), x3ref, N);
  k_hinge_fuse<<<cd((long)N * DF, 256), 256, 0, stream>>>(xl4, hb1, mnB, scB, F(26), x4ref, N);

  // ===== Graph heads (logits averaged into d_out) =====
  auto pred = [&](const float* zref, int n, const int* seg, int pb) {
    k_fillf<<<cd(GNUM * DF, 256), 256, 0, stream>>>(segsum, GNUM * DF, 0.0f);
    k_fillf<<<1, 256, 0, stream>>>(segcnt, GNUM, 0.0f);
    k_seg_count<<<cd(n, 256), 256, 0, stream>>>(seg, segcnt, n);
    k_seg_accum<<<cd((long)n * DF, 256), 256, 0, stream>>>(zref, seg, segsum, (long)n * DF);
    k_seg_div<<<cd(GNUM * DF, 256), 256, 0, stream>>>(segsum, segcnt, segm, GNUM * DF);
    k_cvt<<<cd(DF * 64, 256), 256, 0, stream>>>(F(pb + 1), wb16, DF * 64);         // l1.w
    k_gemm_bf16<4><<<1, 256, 0, stream>>>(segm, wb16, F(pb + 0), z1, GNUM, DF, 1); // relu
    k_mlp2_acc<<<cd(GNUM * CNUM, 256), 256, 0, stream>>>(z1, F(pb + 3), F(pb + 2), outF, GNUM);
  };
  pred(x1ref, N,  batch, 38);
  pred(x2ref, K2, b2,    42);
  pred(x3ref, N,  batch, 46);

  // ===== Cycle-consistency losses =====
  k_fillf<<<1, 256, 0, stream>>>(accs, 4, 0.0f);
  gemm128(x3ref, F(51), F(50), tmpN, N);
  k_mse<<<1024, 256, 0, stream>>>(tmpN, x1ref, &accs[0], (long)N * DF);   // cyc_up
  gemm128(x4ref, F(57), F(56), tmpN, N);
  k_mse<<<1024, 256, 0, stream>>>(tmpN, x6ref, &accs[1], (long)N * DF);   // cyc_lo
  gemm128(x6ref, F(53), F(52), tmpN, N);
  k_mse<<<1024, 256, 0, stream>>>(tmpN, x1ref, &accs[2], (long)N * DF);   // cyc_cr a
  gemm128(x1ref, F(55), F(54), tmpN, N);
  k_mse<<<1024, 256, 0, stream>>>(tmpN, x6ref, &accs[3], (long)N * DF);   // cyc_cr b
  k_final<<<1, 1, 0, stream>>>(accs, outF, (float)N * (float)DF);
}
